// MAGCN_4406636446376
// MI455X (gfx1250) — matrix operand order repaired and checked by reading, not compiled, not run on hardware
//
#include <hip/hip_runtime.h>
#include <math.h>

typedef __attribute__((ext_vector_type(16))) _Float16 v16h;
typedef __attribute__((ext_vector_type(8)))  float    v8f;

#define B_  32
#define L_  200
#define H_  64
#define NH_ 2
#define HD_ 32
#define M_  (B_*L_)    // 6400 rows
#define LP_ 208        // L padded to 13 chunks of 16
#define NEGV (-4294967295.0f)

__device__ __forceinline__ v8f wmma_f16(const v16h& a, const v16h& b, const v8f& c) {
    // D = A(16x32 f16) * B(32x16 f16) + C(16x16 f32)
    return __builtin_amdgcn_wmma_f32_16x16x32_f16(false, a, false, b, (short)0, c, false, false);
}

// ---------------- embedding gather + padding mask ----------------
__global__ void embed_k(const int* __restrict__ ids, const float* __restrict__ emb,
                        float* __restrict__ out) {
    int idx = blockIdx.x * blockDim.x + threadIdx.x;
    if (idx >= M_ * H_) return;
    int m = idx >> 6, j = idx & 63;
    int id = ids[m];
    out[idx] = (id != 0) ? emb[id * H_ + j] : 0.f;
}

// ---------------- layernorm (optionally Y = LN(X + X2)) ----------------
__global__ void ln_k(const float* __restrict__ X, const float* __restrict__ X2,
                     const float* __restrict__ g, const float* __restrict__ bb,
                     float* __restrict__ Y) {
    int row  = blockIdx.x * 8 + (threadIdx.x >> 5);   // one wave32 per row
    int lane = threadIdx.x & 31;
    float a0 = X[row * H_ + lane], a1 = X[row * H_ + lane + 32];
    if (X2) { a0 += X2[row * H_ + lane]; a1 += X2[row * H_ + lane + 32]; }
    float s = a0 + a1;
    for (int o = 16; o >= 1; o >>= 1) s += __shfl_xor(s, o, 32);
    float mu = s * (1.f / 64.f);
    float d0 = a0 - mu, d1 = a1 - mu;
    float v = d0 * d0 + d1 * d1;
    for (int o = 16; o >= 1; o >>= 1) v += __shfl_xor(v, o, 32);
    float inv = rsqrtf(v * (1.f / 64.f) + 1e-8f);
    Y[row * H_ + lane]      = d0 * inv * g[lane]      + bb[lane];
    Y[row * H_ + lane + 32] = d1 * inv * g[lane + 32] + bb[lane + 32];
}

// ---------------- WMMA GEMM: Y[m,n] = epi( sum_k X[m,k]*W[n,k] + bias[n] ) ----------------
// epilogue options: +posAdd[(m%L),n], relu, +resid[m,n], zero row if keepIds[m]==0
__global__ void __launch_bounds__(256) gemm_k(
        const float* __restrict__ X, const float* __restrict__ W,
        const float* __restrict__ bias, const float* __restrict__ posAdd,
        const float* __restrict__ resid, const int* __restrict__ keepIds,
        int relu, float* __restrict__ Y) {
    __shared__ _Float16 Xs[32 * 64];
    __shared__ _Float16 Ws[64 * 64];
    __shared__ float    bs[64];
    int tid = threadIdx.x;
    int m0  = blockIdx.x * 32;
    for (int e = tid; e < 32 * 64; e += 256)
        Xs[e] = (_Float16)X[(m0 + (e >> 6)) * 64 + (e & 63)];
    for (int e = tid; e < 64 * 64; e += 256)
        Ws[e] = (_Float16)W[e];
    if (tid < 64) bs[tid] = bias[tid];
    __syncthreads();

    int w = tid >> 5, lane = tid & 31;
    int mt = w >> 2, nt = w & 3;            // 2 M-tiles x 4 N-tiles, 8 waves
    int lhalf = lane >> 4, l15 = lane & 15;
    v8f acc = {};
    for (int kc = 0; kc < 2; ++kc) {        // K = 64 = 2 x 32
        v16h a, bf;
        int mrow = mt * 16 + l15;
        int kbA = lhalf * 16 + kc * 32;     // A layout: lane m=l15, lanes16-31 hold K hi-half
        int kbB = lhalf * 16 + kc * 32;     // B layout: lane n=l15, lanes16-31 hold K hi-half
        int nb  = nt * 16 + l15;
        #pragma unroll
        for (int j = 0; j < 16; ++j) {
            a[j]  = Xs[mrow * 64 + kbA + j];
            bf[j] = Ws[nb * 64 + kbB + j];
        }
        acc = wmma_f16(a, bf, acc);
    }
    int n = nt * 16 + l15;
    #pragma unroll
    for (int r = 0; r < 8; ++r) {           // D layout: row = r + 8*lhalf
        int m = m0 + mt * 16 + lhalf * 8 + r;
        float v = acc[r] + bs[n];
        if (posAdd) v += posAdd[(m % L_) * 64 + n];
        if (relu)   v = fmaxf(v, 0.f);
        if (resid)  v += resid[m * 64 + n];
        if (keepIds && keepIds[m] == 0) v = 0.f;
        Y[m * 64 + n] = v;
    }
}

// ---------------- fused relative attention per (b, head, 16-row q tile) ----------------
__global__ void __launch_bounds__(256) attn_k(
        const float* __restrict__ qb, const float* __restrict__ kb,
        const float* __restrict__ vb,
        const float* __restrict__ tmk, const float* __restrict__ tmv,
        const float* __restrict__ dmk, const float* __restrict__ dmv,
        const int* __restrict__ tmat, const int* __restrict__ dmat,
        const int* __restrict__ ids, float* __restrict__ out) {
    __shared__ _Float16 ksh[LP_ * HD_];    // k' head slice (f16 for WMMA B)
    __shared__ _Float16 vsh[LP_ * HD_];    // v' head slice
    __shared__ float    qsh[16 * HD_];     // q tile (f32 for VALU dots)
    __shared__ float    sch[16 * LP_];     // scores / attn
    int tid = threadIdx.x;
    int qt = blockIdx.x, h = blockIdx.y, b = blockIdx.z;
    const int hoff = h * HD_;

    for (int e = tid; e < LP_ * HD_; e += 256) {
        int k = e >> 5, d = e & 31;
        float kv = 0.f, vv = 0.f;
        if (k < L_) {
            kv = kb[(b * L_ + k) * H_ + hoff + d];
            vv = vb[(b * L_ + k) * H_ + hoff + d];
        }
        ksh[e] = (_Float16)kv; vsh[e] = (_Float16)vv;
    }
    for (int e = tid; e < 16 * HD_; e += 256) {
        int r = e >> 5, d = e & 31;
        int gq = qt * 16 + r;
        qsh[e] = (gq < L_) ? qb[(b * L_ + gq) * H_ + hoff + d] : 0.f;
    }
    __syncthreads();

    // scores[16 x 208] = q (16x32) x k'^T via WMMA, 13 key chunks over 8 waves
    int w = tid >> 5, lane = tid & 31;
    int lhalf = lane >> 4, l15 = lane & 15;
    v16h a;
    {
        int kbA = lhalf * 16;
        #pragma unroll
        for (int j = 0; j < 16; ++j)
            a[j] = (_Float16)qsh[l15 * HD_ + kbA + j];
    }
    for (int c = w; c < 13; c += 8) {
        v16h bf;
        int kbB = lhalf * 16;
        #pragma unroll
        for (int j = 0; j < 16; ++j)
            bf[j] = ksh[(c * 16 + l15) * HD_ + kbB + j];
        v8f acc = {};
        acc = wmma_f16(a, bf, acc);
        #pragma unroll
        for (int r = 0; r < 8; ++r)
            sch[(lhalf * 8 + r) * LP_ + c * 16 + l15] = acc[r];
    }
    __syncthreads();

    // + q.(tK + dK) gathered from small L2-resident tables (b128 reads), scale, masks
    for (int p = tid; p < 16 * LP_; p += 256) {
        int m = p / LP_, kk = p % LP_;
        int gq = qt * 16 + m;
        bool valid = (gq < L_) && (kk < L_) && (kk <= gq) && (ids[b * L_ + gq] != 0);
        if (valid) {
            int t  = tmat[(b * L_ + gq) * L_ + kk];
            int dd = dmat[(b * L_ + gq) * L_ + kk];
            const float4* tr4 = (const float4*)(tmk + t  * H_ + hoff);
            const float4* dr4 = (const float4*)(dmk + dd * H_ + hoff);
            float accs = sch[p];
            #pragma unroll
            for (int qd = 0; qd < 8; ++qd) {
                float4 tv = tr4[qd], dv = dr4[qd];
                int d0 = qd * 4;
                accs += qsh[m * HD_ + d0 + 0] * (tv.x + dv.x);
                accs += qsh[m * HD_ + d0 + 1] * (tv.y + dv.y);
                accs += qsh[m * HD_ + d0 + 2] * (tv.z + dv.z);
                accs += qsh[m * HD_ + d0 + 3] * (tv.w + dv.w);
            }
            sch[p] = accs * 0.1767766953f;   // 1/sqrt(32)
        } else {
            sch[p] = NEGV;
        }
    }
    __syncthreads();

    // softmax: 16 threads per row (rows contiguous within waves)
    {
        int m = tid >> 4, j = tid & 15;
        float mx = -3.4e38f;
        for (int kk = j; kk < LP_; kk += 16) mx = fmaxf(mx, sch[m * LP_ + kk]);
        for (int o = 8; o >= 1; o >>= 1) mx = fmaxf(mx, __shfl_xor(mx, o, 16));
        float sum = 0.f;
        for (int kk = j; kk < LP_; kk += 16) {
            float e = __expf(sch[m * LP_ + kk] - mx);
            sch[m * LP_ + kk] = e; sum += e;
        }
        for (int o = 8; o >= 1; o >>= 1) sum += __shfl_xor(sum, o, 16);
        float inv = 1.f / sum;
        for (int kk = j; kk < LP_; kk += 16) sch[m * LP_ + kk] *= inv;
    }
    __syncthreads();

    // out[m,:] = sum_k attn * (v'[k,:] + tV[t,:] + dV[dd,:]); causal => k <= gq.
    // 16 threads per row; each owns all 32 dims in registers, strides over keys,
    // indices loaded once per key, tables read as 128-bit vectors, then a
    // lane-permute (shfl_xor) tree reduction across the 16 lanes.
    {
        int m = tid >> 4, j = tid & 15;
        int gq = qt * 16 + m;
        float acc[32];
        #pragma unroll
        for (int d = 0; d < 32; ++d) acc[d] = 0.f;
        if (gq < L_) {
            const int* tr = tmat + (b * L_ + gq) * L_;
            const int* dr = dmat + (b * L_ + gq) * L_;
            __builtin_prefetch(tr, 0, 1);
            __builtin_prefetch(dr, 0, 1);
            for (int kk = j; kk <= gq; kk += 16) {
                float aw = sch[m * LP_ + kk];
                int t = tr[kk], dd = dr[kk];
                const float4* tv4 = (const float4*)(tmv + t  * H_ + hoff);
                const float4* dv4 = (const float4*)(dmv + dd * H_ + hoff);
                #pragma unroll
                for (int qd = 0; qd < 8; ++qd) {
                    float4 tv = tv4[qd], dv = dv4[qd];
                    int d0 = qd * 4;
                    acc[d0 + 0] += aw * ((float)vsh[kk * HD_ + d0 + 0] + tv.x + dv.x);
                    acc[d0 + 1] += aw * ((float)vsh[kk * HD_ + d0 + 1] + tv.y + dv.y);
                    acc[d0 + 2] += aw * ((float)vsh[kk * HD_ + d0 + 2] + tv.z + dv.z);
                    acc[d0 + 3] += aw * ((float)vsh[kk * HD_ + d0 + 3] + tv.w + dv.w);
                }
            }
        }
        #pragma unroll
        for (int d = 0; d < 32; ++d) {
            #pragma unroll
            for (int o = 8; o >= 1; o >>= 1)
                acc[d] += __shfl_xor(acc[d], o, 16);
        }
        if (gq < L_ && j == 0) {
            float4* op = (float4*)(out + (b * L_ + gq) * H_ + hoff);
            #pragma unroll
            for (int qd = 0; qd < 8; ++qd)
                op[qd] = float4{acc[qd * 4], acc[qd * 4 + 1], acc[qd * 4 + 2], acc[qd * 4 + 3]};
        }
    }
}

extern "C" void kernel_launch(void* const* d_in, const int* in_sizes, int n_in,
                              void* d_out, int out_size, void* d_ws, size_t ws_size,
                              hipStream_t stream) {
    const int*   log_seqs = (const int*)d_in[0];
    const int*   tmat = (const int*)d_in[1];
    const int*   dmat = (const int*)d_in[2];
    const float* item_emb = (const float*)d_in[3];
    const float* posK = (const float*)d_in[4];
    const float* posV = (const float*)d_in[5];
    const float* tmk  = (const float*)d_in[6];
    const float* tmv  = (const float*)d_in[7];
    const float* dmk  = (const float*)d_in[8];
    const float* dmv  = (const float*)d_in[9];
    const float* Wq = (const float*)d_in[10];
    const float* bq = (const float*)d_in[11];
    const float* Wk = (const float*)d_in[12];
    const float* bk = (const float*)d_in[13];
    const float* Wv = (const float*)d_in[14];
    const float* bv = (const float*)d_in[15];
    const float* ln1g = (const float*)d_in[16];
    const float* ln1b = (const float*)d_in[17];
    const float* ln2g = (const float*)d_in[18];
    const float* ln2b = (const float*)d_in[19];
    const float* c1w = (const float*)d_in[20];
    const float* c1b = (const float*)d_in[21];
    const float* c2w = (const float*)d_in[22];
    const float* c2b = (const float*)d_in[23];
    const float* lnfg = (const float*)d_in[24];
    const float* lnfb = (const float*)d_in[25];

    const size_t SZ = (size_t)M_ * H_;      // 409600 floats per activation buffer
    float* ws   = (float*)d_ws;
    float* seqs = ws + 0 * SZ;
    float* Qb   = ws + 1 * SZ;
    float* qbuf = ws + 2 * SZ;
    float* kbuf = ws + 3 * SZ;
    float* vbuf = ws + 4 * SZ;
    float* attb = ws + 5 * SZ;
    float* ffh  = ws + 6 * SZ;
    float* s2   = ws + 7 * SZ;

    embed_k<<<(M_ * H_ + 255) / 256, 256, 0, stream>>>(log_seqs, item_emb, seqs);
    for (int i = 0; i < 2; ++i) {
        ln_k<<<M_ / 8, 256, 0, stream>>>(seqs, nullptr, ln1g + i * H_, ln1b + i * H_, Qb);
        gemm_k<<<M_ / 32, 256, 0, stream>>>(Qb,   Wq + i * H_ * H_, bq + i * H_, nullptr, nullptr, nullptr, 0, qbuf);
        gemm_k<<<M_ / 32, 256, 0, stream>>>(seqs, Wk + i * H_ * H_, bk + i * H_, posK,    nullptr, nullptr, 0, kbuf);
        gemm_k<<<M_ / 32, 256, 0, stream>>>(seqs, Wv + i * H_ * H_, bv + i * H_, posV,    nullptr, nullptr, 0, vbuf);
        attn_k<<<dim3(13, NH_, B_), 256, 0, stream>>>(qbuf, kbuf, vbuf, tmk, tmv, dmk, dmv,
                                                      tmat, dmat, log_seqs, attb);
        ln_k<<<M_ / 8, 256, 0, stream>>>(Qb, attb, ln2g + i * H_, ln2b + i * H_, s2);
        gemm_k<<<M_ / 32, 256, 0, stream>>>(s2,  c1w + i * H_ * H_, c1b + i * H_, nullptr, nullptr, nullptr, 1, ffh);
        gemm_k<<<M_ / 32, 256, 0, stream>>>(ffh, c2w + i * H_ * H_, c2b + i * H_, nullptr, s2, log_seqs, 0, seqs);
    }
    ln_k<<<M_ / 8, 256, 0, stream>>>(seqs, nullptr, lnfg, lnfb, (float*)d_out);
}
